// _DeepSeekMoEFFN_22797686407762
// MI455X (gfx1250) — matrix operand order repaired and checked by reading, not compiled, not run on hardware
//
#include <hip/hip_runtime.h>

// ---------------------------------------------------------------------------
// DeepSeek MoE FFN for gfx1250 (MI455X): bf16 WMMA + Tensor-Data-Mover path.
// T=4096 tokens, H=1024, I=512, E=8 routed experts (top-2), shared SI=1024.
//
// Fusion: routing weight folded into the scaled SwiGLU hidden state so the
// expert combine + down proj + shared down proj become ONE GEMM with K=5120.
// GEMMs stage A/B tiles into LDS with TENSOR_LOAD_TO_LDS (double-buffered,
// TENSORcnt-synchronized), WMMAs consume fragments from LDS.
// ---------------------------------------------------------------------------

typedef unsigned short u16;
typedef __attribute__((ext_vector_type(8)))  __bf16 v8bf;
typedef __attribute__((ext_vector_type(16))) __bf16 v16bf;
typedef __attribute__((ext_vector_type(8)))  float  v8f;
typedef __attribute__((ext_vector_type(4)))  unsigned int v4u;
typedef __attribute__((ext_vector_type(4)))  int v4i;
typedef __attribute__((ext_vector_type(8)))  int v8i;

#define T_TOK 4096
#define H_DIM 1024
#define I_DIM 512
#define N_E   8
#define NEB   10           // 8 routed expert blocks + 2 shared blocks (width 512)
#define KD    (NEB * 512)  // 5120 : fused K for the down GEMM

// K-chunking for TDM-staged GEMMs
#define KC    64           // K elements per staged chunk (128 B rows)
#define KCP   80           // padded LDS row pitch in elements (128 B + 32 B pad)
#define PADI  4            // pad_interval code: 32 DWORDs = 128 B
#define PADA  7            // pad_amount  code: 8 DWORDs  = 32 B

__device__ __forceinline__ u16 f2bf_bits(float f) {
  unsigned u = __float_as_uint(f);
  u += 0x7FFFu + ((u >> 16) & 1u);   // round-to-nearest-even
  return (u16)(u >> 16);
}

// A-fragment (16x32 bf16): lane half lh holds K = [16lh,16lh+16)
__device__ __forceinline__ v16bf load_a_frag(const u16* row, int k, int lh) {
  v8bf lo = *(const v8bf*)(row + k + 16 * lh);
  v8bf hi = *(const v8bf*)(row + k + 16 * lh + 8);
  return __builtin_shufflevector(lo, hi, 0, 1, 2, 3, 4, 5, 6, 7,
                                 8, 9, 10, 11, 12, 13, 14, 15);
}

// ---------------------------------------------------------------------------
// TDM: issue one 2D tile load (tile_rows x KC bf16 elements, global row
// stride row_stride elements) into LDS at lds_off, with 32B row padding.
// Descriptor bitfields per CDNA5 ISA 8.3/8.4 (D# group0/group1).
// ---------------------------------------------------------------------------
__device__ __forceinline__ void tdm_load_2d(unsigned lds_off, const u16* gptr,
                                            int tile_rows, int row_stride) {
  unsigned long long ga = (unsigned long long)(uintptr_t)gptr;
  v4u g0;
  g0.x = 1u;                                    // count=1, user descriptor
  g0.y = lds_off;                               // lds_addr [63:32]
  g0.z = (unsigned)(ga & 0xFFFFFFFFu);          // global_addr lo
  g0.w = (unsigned)(ga >> 32) | 0x80000000u;    // global_addr hi | type=2 @[127:126]

  v8i g1;
  g1[0] = (int)((1u << 16)        // data_size = 1 -> 2 bytes
              | (1u << 20)        // pad_enable
              | ((unsigned)PADI << 22)
              | ((unsigned)PADA << 25));
  g1[1] = (int)(((unsigned)KC & 0xFFFFu) << 16);                 // tensor_dim0 lo16
  g1[2] = (int)(((unsigned)KC >> 16) |
                (((unsigned)tile_rows & 0xFFFFu) << 16));        // td0 hi | td1 lo
  g1[3] = (int)((((unsigned)tile_rows >> 16) & 0xFFFFu) |
                ((unsigned)KC << 16));                           // td1 hi | tile_dim0
  g1[4] = (int)((unsigned)tile_rows & 0xFFFFu);                  // tile_dim1 | tile_dim2=0
  g1[5] = (int)(unsigned)row_stride;                             // tensor_dim0_stride lo32
  g1[6] = 0;
  g1[7] = 0;

  v4i z4 = {0, 0, 0, 0};
#if defined(__clang_major__) && (__clang_major__ >= 23)
  v8i z8 = {0, 0, 0, 0, 0, 0, 0, 0};
  __builtin_amdgcn_tensor_load_to_lds(g0, g1, z4, z4, z8, 0);
#else
  __builtin_amdgcn_tensor_load_to_lds(g0, g1, z4, z4, 0);
#endif
}

__device__ __forceinline__ unsigned lds_offset_of(const void* p) {
  // Low 32 bits of the flat address of an LDS object are its LDS byte address.
  return (unsigned)(uintptr_t)p;
}

// ---------------------------------------------------------------------------
// fp32 -> bf16 elementwise (activations)
// ---------------------------------------------------------------------------
__global__ __launch_bounds__(256) void k_f32_to_bf16(const float* __restrict__ in,
                                                     u16* __restrict__ out, int n) {
  int i = blockIdx.x * 256 + threadIdx.x;
  if (i < n) out[i] = f2bf_bits(in[i]);
}

// ---------------------------------------------------------------------------
// Transpose + convert: src fp32 [R][C] (row-major, batched) ->
//   dst bf16 [C][R] with dst row stride ldd (batched by dstBatch elements).
// ---------------------------------------------------------------------------
__global__ __launch_bounds__(256) void k_transpose_bf16(
    const float* __restrict__ src, long srcBatch, int R, int C,
    u16* __restrict__ dst, long dstBatch, int ldd) {
  __shared__ float tile[32][33];
  const float* s = src + (size_t)blockIdx.z * srcBatch;
  u16* d = dst + (size_t)blockIdx.z * dstBatch;

  int x = blockIdx.x * 32 + threadIdx.x;              // src col
#pragma unroll
  for (int j = 0; j < 32; j += 8) {
    int y = blockIdx.y * 32 + threadIdx.y + j;        // src row
    tile[threadIdx.y + j][threadIdx.x] = s[(size_t)y * C + x];
  }
  __syncthreads();
  int x2 = blockIdx.y * 32 + threadIdx.x;             // dst col = src row
#pragma unroll
  for (int j = 0; j < 32; j += 8) {
    int y2 = blockIdx.x * 32 + threadIdx.y + j;       // dst row = src col
    d[(size_t)y2 * ldd + x2] = f2bf_bits(tile[threadIdx.x][threadIdx.y + j]);
  }
}

// ---------------------------------------------------------------------------
// Routing: one wave32 per token. logits = x[t]@gate_w, softmax, top-2,
// renormalize; emit dense cw[T][10] (cols 8,9 = 1.0 for shared blocks).
// ---------------------------------------------------------------------------
__global__ __launch_bounds__(256) void k_route(const float* __restrict__ x,
                                               const float* __restrict__ gw,
                                               float* __restrict__ cw) {
  int lane = threadIdx.x & 31;
  int wave = threadIdx.x >> 5;
  int t = blockIdx.x * 8 + wave;
  const float* xr = x + (size_t)t * H_DIM;

  float acc[N_E] = {0.f, 0.f, 0.f, 0.f, 0.f, 0.f, 0.f, 0.f};
  for (int h = lane; h < H_DIM; h += 32) {
    float xv = xr[h];
    const float4* g4 = (const float4*)(gw + (size_t)h * N_E);
    float4 a = g4[0], b = g4[1];
    acc[0] += xv * a.x; acc[1] += xv * a.y; acc[2] += xv * a.z; acc[3] += xv * a.w;
    acc[4] += xv * b.x; acc[5] += xv * b.y; acc[6] += xv * b.z; acc[7] += xv * b.w;
  }
#pragma unroll
  for (int m = 16; m > 0; m >>= 1)
#pragma unroll
    for (int e = 0; e < N_E; ++e) acc[e] += __shfl_xor(acc[e], m, 32);

  if (lane == 0) {
    float mx = acc[0];
#pragma unroll
    for (int e = 1; e < N_E; ++e) mx = fmaxf(mx, acc[e]);
    float s[N_E], sum = 0.f;
#pragma unroll
    for (int e = 0; e < N_E; ++e) { s[e] = __expf(acc[e] - mx); sum += s[e]; }
    float inv = 1.f / sum;
    int i0 = 0, i1 = -1;
    float b0 = -1.f, b1 = -1.f;
#pragma unroll
    for (int e = 0; e < N_E; ++e) {
      float v = s[e] * inv;
      if (v > b0)      { b1 = b0; i1 = i0; b0 = v; i0 = e; }
      else if (v > b1) { b1 = v; i1 = e; }
    }
    float wnorm = 1.f / (b0 + b1);
    float* cwt = cw + (size_t)t * NEB;
#pragma unroll
    for (int e = 0; e < N_E; ++e) {
      float v = 0.f;
      if (e == i0) v = b0 * wnorm;
      else if (e == i1) v = b1 * wnorm;
      cwt[e] = v;
    }
    cwt[8] = 1.f;
    cwt[9] = 1.f;
  }
}

// ---------------------------------------------------------------------------
// Pass A: per expert block eb (10 blocks, width 512):
//   g = Xbf @ W1t[eb]^T, u = Xbf @ W2t[eb]^T   (K = H = 1024)
//   hout[t, eb*512+i] = cw[t,eb] * silu(g) * u  (bf16)
// Block tile M=64, N=128; 8 waves (4 along M x 2 along N), wave tile 16x64.
// A/B1/B2 K-chunks staged via TDM into double-buffered LDS.
// ---------------------------------------------------------------------------
__global__ __launch_bounds__(256) void k_up(const u16* __restrict__ xb,    // [T][1024]
                                            const u16* __restrict__ w1t,   // [10][512][1024]
                                            const u16* __restrict__ w2t,   // [10][512][1024]
                                            const float* __restrict__ cw,  // [T][10]
                                            u16* __restrict__ hout) {      // [T][5120]
  __shared__ __align__(32) u16 sA [2][64  * KCP];   // 20,480 B
  __shared__ __align__(32) u16 sB1[2][128 * KCP];   // 40,960 B
  __shared__ __align__(32) u16 sB2[2][128 * KCP];   // 40,960 B

  const int lane = threadIdx.x & 31;
  const int wave = threadIdx.x >> 5;
  const int wm = wave & 3, wn = wave >> 2;
  const int eb = blockIdx.z;
  const int m0 = blockIdx.x * 64;
  const int nE0 = blockIdx.y * 128;                 // col base within expert block
  const int lr = lane & 15, lh = lane >> 4;

  const u16* Abase  = xb  + (size_t)m0 * H_DIM;
  const u16* B1base = w1t + ((size_t)eb * 512 + nE0) * H_DIM;
  const u16* B2base = w2t + ((size_t)eb * 512 + nE0) * H_DIM;

  v8f accg[4] = {v8f{}, v8f{}, v8f{}, v8f{}};
  v8f accu[4] = {v8f{}, v8f{}, v8f{}, v8f{}};

  const int NCH = H_DIM / KC;                       // 16 chunks

  if (threadIdx.x < 32) {                           // wave 0 drives the TDM
    tdm_load_2d(lds_offset_of(&sA [0][0]), Abase,  64,  H_DIM);
    tdm_load_2d(lds_offset_of(&sB1[0][0]), B1base, 128, H_DIM);
    tdm_load_2d(lds_offset_of(&sB2[0][0]), B2base, 128, H_DIM);
    __builtin_amdgcn_s_wait_tensorcnt(0);
  }
  __syncthreads();

  for (int c = 0; c < NCH; ++c) {
    const int buf = c & 1;
    if (threadIdx.x < 32 && c + 1 < NCH) {          // prefetch next chunk
      const int nb = buf ^ 1;
      tdm_load_2d(lds_offset_of(&sA [nb][0]), Abase  + (c + 1) * KC, 64,  H_DIM);
      tdm_load_2d(lds_offset_of(&sB1[nb][0]), B1base + (c + 1) * KC, 128, H_DIM);
      tdm_load_2d(lds_offset_of(&sB2[nb][0]), B2base + (c + 1) * KC, 128, H_DIM);
    }

    const u16* As = &sA[buf][(wm * 16 + lr) * KCP];
#pragma unroll
    for (int kk = 0; kk < KC; kk += 32) {
      v16bf a = load_a_frag(As, kk, lh);
#pragma unroll
      for (int j = 0; j < 4; ++j) {
        const int brow = (wn * 64 + j * 16 + lr) * KCP + kk + 16 * lh;
        v16bf b1 = *(const v16bf*)(&sB1[buf][brow]);
        accg[j] = __builtin_amdgcn_wmma_f32_16x16x32_bf16(
            false, a, false, b1, (short)0, accg[j], false, false);
        v16bf b2 = *(const v16bf*)(&sB2[buf][brow]);
        accu[j] = __builtin_amdgcn_wmma_f32_16x16x32_bf16(
            false, a, false, b2, (short)0, accu[j], false, false);
      }
    }

    __syncthreads();                                // all waves done with buf
    if (threadIdx.x < 32 && c + 1 < NCH)
      __builtin_amdgcn_s_wait_tensorcnt(0);         // next buffer landed
    __syncthreads();
  }

  float cws[8];
#pragma unroll
  for (int r = 0; r < 8; ++r)
    cws[r] = cw[(size_t)(m0 + wm * 16 + r + lh * 8) * NEB + eb];

#pragma unroll
  for (int j = 0; j < 4; ++j) {
    const int col = eb * 512 + nE0 + wn * 64 + j * 16 + lr;
#pragma unroll
    for (int r = 0; r < 8; ++r) {
      const int m = m0 + wm * 16 + r + lh * 8;
      float g = accg[j][r];
      float u = accu[j][r];
      float h = (g / (1.f + __expf(-g))) * u * cws[r];   // silu(g)*u*cw
      hout[(size_t)m * KD + col] = f2bf_bits(h);
    }
  }
}

// ---------------------------------------------------------------------------
// Pass B: out[T][1024] = H_all[T][5120] @ Wd_all^T  (wdt is [1024][5120]).
// Routed combine + shared expert in one GEMM. fp32 straight into d_out.
// ---------------------------------------------------------------------------
__global__ __launch_bounds__(256) void k_down(const u16* __restrict__ hb,   // [T][5120]
                                              const u16* __restrict__ wdt,  // [1024][5120]
                                              float* __restrict__ out) {    // [T][1024]
  __shared__ __align__(32) u16 sA[2][64  * KCP];    // 20,480 B
  __shared__ __align__(32) u16 sB[2][128 * KCP];    // 40,960 B

  const int lane = threadIdx.x & 31;
  const int wave = threadIdx.x >> 5;
  const int wm = wave & 3, wn = wave >> 2;
  const int m0 = blockIdx.x * 64;
  const int n0 = blockIdx.y * 128;
  const int lr = lane & 15, lh = lane >> 4;

  const u16* Abase = hb  + (size_t)m0 * KD;
  const u16* Bbase = wdt + (size_t)n0 * KD;

  v8f acc[4] = {v8f{}, v8f{}, v8f{}, v8f{}};

  const int NCH = KD / KC;                          // 80 chunks

  if (threadIdx.x < 32) {
    tdm_load_2d(lds_offset_of(&sA[0][0]), Abase, 64,  KD);
    tdm_load_2d(lds_offset_of(&sB[0][0]), Bbase, 128, KD);
    __builtin_amdgcn_s_wait_tensorcnt(0);
  }
  __syncthreads();

  for (int c = 0; c < NCH; ++c) {
    const int buf = c & 1;
    if (threadIdx.x < 32 && c + 1 < NCH) {
      const int nb = buf ^ 1;
      tdm_load_2d(lds_offset_of(&sA[nb][0]), Abase + (c + 1) * KC, 64,  KD);
      tdm_load_2d(lds_offset_of(&sB[nb][0]), Bbase + (c + 1) * KC, 128, KD);
    }

    const u16* As = &sA[buf][(wm * 16 + lr) * KCP];
#pragma unroll
    for (int kk = 0; kk < KC; kk += 32) {
      v16bf a = load_a_frag(As, kk, lh);
#pragma unroll
      for (int j = 0; j < 4; ++j) {
        v16bf b = *(const v16bf*)(&sB[buf][(wn * 64 + j * 16 + lr) * KCP + kk + 16 * lh]);
        acc[j] = __builtin_amdgcn_wmma_f32_16x16x32_bf16(
            false, a, false, b, (short)0, acc[j], false, false);
      }
    }

    __syncthreads();
    if (threadIdx.x < 32 && c + 1 < NCH)
      __builtin_amdgcn_s_wait_tensorcnt(0);
    __syncthreads();
  }

#pragma unroll
  for (int j = 0; j < 4; ++j) {
    const int col = n0 + wn * 64 + j * 16 + lr;
#pragma unroll
    for (int r = 0; r < 8; ++r) {
      const int m = m0 + wm * 16 + r + lh * 8;
      out[(size_t)m * H_DIM + col] = acc[j][r];
    }
  }
}

// ---------------------------------------------------------------------------
// Host-side launcher
// ---------------------------------------------------------------------------
extern "C" void kernel_launch(void* const* d_in, const int* in_sizes, int n_in,
                              void* d_out, int out_size, void* d_ws, size_t ws_size,
                              hipStream_t stream) {
  const float* x      = (const float*)d_in[0];  // [2,2048,1024] -> [4096,1024]
  const float* gate_w = (const float*)d_in[1];  // [1024,8]
  const float* wg     = (const float*)d_in[2];  // [8,1024,512]
  const float* wu     = (const float*)d_in[3];  // [8,1024,512]
  const float* wd     = (const float*)d_in[4];  // [8,512,1024]
  const float* sg     = (const float*)d_in[5];  // [1024,1024]
  const float* su     = (const float*)d_in[6];  // [1024,1024]
  const float* sd     = (const float*)d_in[7];  // [1024,1024]
  float* out = (float*)d_out;

  // Workspace carve-up (all chunks 256B-aligned by construction): ~82 MB total
  char* wsb = (char*)d_ws;
  size_t off = 0;
  u16* ws_x   = (u16*)(wsb + off); off += (size_t)T_TOK * H_DIM * 2;       //  8 MB
  u16* ws_w1t = (u16*)(wsb + off); off += (size_t)NEB * 512 * H_DIM * 2;   // 10 MB
  u16* ws_w2t = (u16*)(wsb + off); off += (size_t)NEB * 512 * H_DIM * 2;   // 10 MB
  u16* ws_wdt = (u16*)(wsb + off); off += (size_t)H_DIM * KD * 2;          // 10 MB
  u16* ws_h   = (u16*)(wsb + off); off += (size_t)T_TOK * KD * 2;          // 40 MB
  float* ws_cw = (float*)(wsb + off); off += (size_t)T_TOK * NEB * 4;      // 160 KB
  (void)ws_size; (void)in_sizes; (void)n_in; (void)out_size;

  // 1) activations fp32 -> bf16
  {
    int n = T_TOK * H_DIM;
    k_f32_to_bf16<<<(n + 255) / 256, 256, 0, stream>>>(x, ws_x, n);
  }

  // 2) weight transpose+convert into B^T bf16 layouts
  dim3 tb(32, 8);
  k_transpose_bf16<<<dim3(512 / 32, 1024 / 32, N_E), tb, 0, stream>>>(
      wg, (long)H_DIM * I_DIM, H_DIM, I_DIM, ws_w1t, (long)I_DIM * H_DIM, H_DIM);
  k_transpose_bf16<<<dim3(1024 / 32, 1024 / 32, 1), tb, 0, stream>>>(
      sg, 0, H_DIM, 1024, ws_w1t + (size_t)8 * 512 * H_DIM, 0, H_DIM);
  k_transpose_bf16<<<dim3(512 / 32, 1024 / 32, N_E), tb, 0, stream>>>(
      wu, (long)H_DIM * I_DIM, H_DIM, I_DIM, ws_w2t, (long)I_DIM * H_DIM, H_DIM);
  k_transpose_bf16<<<dim3(1024 / 32, 1024 / 32, 1), tb, 0, stream>>>(
      su, 0, H_DIM, 1024, ws_w2t + (size_t)8 * 512 * H_DIM, 0, H_DIM);
  k_transpose_bf16<<<dim3(1024 / 32, 512 / 32, N_E), tb, 0, stream>>>(
      wd, (long)I_DIM * H_DIM, I_DIM, H_DIM, ws_wdt, (long)I_DIM, KD);
  k_transpose_bf16<<<dim3(1024 / 32, 1024 / 32, 1), tb, 0, stream>>>(
      sd, 0, 1024, H_DIM, ws_wdt + 4096, 0, KD);

  // 3) routing -> dense combine weights cw[T][10]
  k_route<<<T_TOK / 8, 256, 0, stream>>>(x, gate_w, ws_cw);

  // 4) pass A: scaled SwiGLU hidden states (bf16 WMMA, TDM-staged)
  k_up<<<dim3(T_TOK / 64, 512 / 128, NEB), 256, 0, stream>>>(
      ws_x, ws_w1t, ws_w2t, ws_cw, ws_h);

  // 5) pass B: single fused down GEMM -> final output (moe + shared)
  k_down<<<dim3(T_TOK / 64, H_DIM / 128), 256, 0, stream>>>(ws_h, ws_wdt, out);
}
